// MHPLSTMCore_21303037788177
// MI455X (gfx1250) — matrix operand order repaired and checked by reading, not compiled, not run on hardware
//
#include <hip/hip_runtime.h>
#include <hip/hip_bf16.h>
#include <math.h>

// ---------- types ----------
typedef __bf16 bf16_t;
typedef __attribute__((ext_vector_type(16))) __bf16   v16bf;
typedef __attribute__((ext_vector_type(8)))  __bf16   v8bf;
typedef __attribute__((ext_vector_type(8)))  float    v8f;
typedef __attribute__((ext_vector_type(4)))  unsigned v4u;
typedef __attribute__((ext_vector_type(8)))  int      v8i;
typedef __attribute__((ext_vector_type(4)))  int      v4i;

// ---------- problem dims ----------
#define B_   8
#define S_   2048
#define H_   8
#define ID_  256
#define OD_  256
#define K2_  512                  // 2*ID = ID+OD (both GEMM K dims)
#define M_   (B_ * S_ * H_)       // 131072 rows
#define N1_  768                  // 3*OD

__device__ __forceinline__ float sigmoidf_(float x) {
  return 1.0f / (1.0f + expf(-x));
}

__device__ __forceinline__ unsigned lds_off_(const void* p) {
  // generic LDS address = {SHARED_BASE, offset32}; low 32 bits are the offset
  return (unsigned)(unsigned long long)p;
}

// ---------- TDM: 2D tile global -> LDS (D# groups packed per CDNA5 ISA 8.3/8.4) ----------
__device__ __forceinline__ void tdm_load_2d(unsigned lds_byte_off, const void* gptr,
                                            unsigned tile_d0, unsigned tile_d1,
                                            unsigned tensor_d0, unsigned tensor_d1,
                                            unsigned long long stride0_elems,
                                            unsigned data_size_code /*1 => 2B*/) {
  unsigned long long ga = (unsigned long long)gptr;
  v4u g0;
  g0[0] = 1u;                                               // count=1, user desc
  g0[1] = lds_byte_off;                                     // lds_addr [63:32]
  g0[2] = (unsigned)(ga & 0xFFFFFFFFull);                   // global_addr [95:64]
  g0[3] = (unsigned)((ga >> 32) & 0x1FFFFFFull) | (2u << 30); // addr msbs + type=2
  v8i g1;
  g1[0] = (int)(data_size_code << 16);                      // wg_mask=0, data_size
  g1[1] = (int)((tensor_d0 & 0xFFFFu) << 16);               // tensor_dim0 lo
  g1[2] = (int)((tensor_d0 >> 16) | ((tensor_d1 & 0xFFFFu) << 16));
  g1[3] = (int)(((tensor_d1 >> 16) & 0xFFFFu) | (tile_d0 << 16));
  g1[4] = (int)(tile_d1 & 0xFFFFu);                         // tile_dim1, tile_dim2=0
  g1[5] = (int)(stride0_elems & 0xFFFFFFFFull);             // tensor_dim0_stride lo
  g1[6] = (int)((stride0_elems >> 32) & 0xFFFFull);         // stride hi, dim1_stride=0
  g1[7] = 0;
  v4i z4 = {0, 0, 0, 0};
#if defined(__clang_major__) && (__clang_major__ >= 23)
  v8i z8 = {0, 0, 0, 0, 0, 0, 0, 0};
  __builtin_amdgcn_tensor_load_to_lds(g0, g1, z4, z4, z8, 0);
#else
  __builtin_amdgcn_tensor_load_to_lds(g0, g1, z4, z4, 0);
#endif
}

// ---------- WMMA fragment loaders (wave32 layouts, CDNA5 ISA 7.12.2) ----------
// A 16x32 bf16: lane L holds row M=L&15; halves 0..7 -> K = (L>=16?8:0)+h,
//               halves 8..15 -> K = 16+(L>=16?8:0)+(h-8)
__device__ __forceinline__ v16bf load_a_frag(const bf16_t* rowPtr, int k0, int hi) {
  v8bf lo = *(const v8bf*)(rowPtr + k0 + hi * 8);
  v8bf hh = *(const v8bf*)(rowPtr + k0 + 16 + hi * 8);
  return __builtin_shufflevector(lo, hh, 0, 1, 2, 3, 4, 5, 6, 7,
                                 8, 9, 10, 11, 12, 13, 14, 15);
}
// B 32x16 bf16: lane L holds col N=L&15; halves 0..7 -> K = (L>=16?8:0)+h,
//               halves 8..15 -> K = 16+(L>=16?8:0)+(h-8)
__device__ __forceinline__ v16bf load_b_frag(const bf16_t* rowPtr, int k0, int hi) {
  v8bf lo = *(const v8bf*)(rowPtr + k0 + hi * 8);
  v8bf hh = *(const v8bf*)(rowPtr + k0 + 16 + hi * 8);
  return __builtin_shufflevector(lo, hh, 0, 1, 2, 3, 4, 5, 6, 7,
                                 8, 9, 10, 11, 12, 13, 14, 15);
}

// ================= K0: weights fp32 -> bf16 =================
__global__ void k_convert_weights(const float* __restrict__ whid,
                                  const float* __restrict__ wog,
                                  bf16_t* __restrict__ whid_bf,
                                  bf16_t* __restrict__ wog_bf) {
  int i = blockIdx.x * blockDim.x + threadIdx.x;
  if (i < N1_ * K2_) whid_bf[i] = (bf16_t)whid[i];
  if (i < OD_ * K2_) wog_bf[i]  = (bf16_t)wog[i];
}

// ================= K1a: cumsum along S; pack x2 = [hin | csum] as bf16 =======
__global__ void k_cumsum_pack(const float* __restrict__ hin,
                              bf16_t* __restrict__ x2) {
  int t = blockIdx.x * blockDim.x + threadIdx.x;   // B*H*ID = 16384 lanes
  int b = t / (H_ * ID_);
  int r = t % (H_ * ID_);
  int h = r / ID_, id = r % ID_;
  float acc = 0.0f;
  for (int s = 0; s < S_; ++s) {
    size_t row = ((size_t)b * S_ + s) * H_ + h;
    float x = hin[row * ID_ + id];
    acc += x;
    x2[row * K2_ + id]       = (bf16_t)x;
    x2[row * K2_ + ID_ + id] = (bf16_t)acc;
  }
}

// ================= K1b: LayerNorm over (H,ID) per (b,s), in-place on x2 high half
__global__ void k_layernorm(bf16_t* __restrict__ x2,
                            const float* __restrict__ ln_w,
                            const float* __restrict__ ln_b) {
  __shared__ float rsum[256];
  __shared__ float rsq[256];
  const int bs = blockIdx.x;                  // b*S + s
  const int tid = threadIdx.x;
  float vals[8];
  float sum = 0.f, sq = 0.f;
  for (int i = 0; i < 8; ++i) {
    int j  = tid + 256 * i;                   // 0..2047 over (h,id)
    int h  = j >> 8, id = j & 255;
    float v = (float)x2[((size_t)bs * H_ + h) * K2_ + ID_ + id];
    vals[i] = v; sum += v; sq += v * v;
  }
  rsum[tid] = sum; rsq[tid] = sq;
  __syncthreads();
  for (int st = 128; st > 0; st >>= 1) {
    if (tid < st) { rsum[tid] += rsum[tid + st]; rsq[tid] += rsq[tid + st]; }
    __syncthreads();
  }
  float mean = rsum[0] * (1.0f / 2048.0f);
  float var  = rsq[0] * (1.0f / 2048.0f) - mean * mean;
  float rstd = rsqrtf(var + 1e-6f);
  for (int i = 0; i < 8; ++i) {
    int j = tid + 256 * i;
    int h = j >> 8, id = j & 255;
    float v = (vals[i] - mean) * rstd * ln_w[j] + ln_b[j];
    x2[((size_t)bs * H_ + h) * K2_ + ID_ + id] = (bf16_t)v;
  }
}

// ================= K2: GEMM1 (TDM-staged, bf16 WMMA) + gate epilogue =========
// block = 128 threads (4 waves). M tile = 64 rows, OD tile = 16 cols x 3 groups.
__global__ void k_gemm1_gates(const bf16_t* __restrict__ x2,
                              const bf16_t* __restrict__ whid,
                              const float* __restrict__ b_hid,
                              float* __restrict__ fgate,
                              float* __restrict__ igh) {
  extern __shared__ char smem_raw[];
  bf16_t* sA = (bf16_t*)smem_raw;                        // 64 x 512 bf16 (64 KiB)
  bf16_t* sB = (bf16_t*)(smem_raw + 64 * K2_ * 2);       // 3 x 16 x 512 (48 KiB)

  const int m0 = blockIdx.x * 64;
  const int n0 = blockIdx.y * 16;

  if (threadIdx.x < 32) {   // one wave issues the DMA descriptors
    tdm_load_2d(lds_off_(sA), x2 + (size_t)m0 * K2_, K2_, 64, K2_, M_, K2_, 1);
    for (int g = 0; g < 3; ++g)
      tdm_load_2d(lds_off_(sB + g * 16 * K2_),
                  whid + (size_t)(g * OD_ + n0) * K2_, K2_, 16, K2_, N1_, K2_, 1);
    __builtin_amdgcn_s_wait_tensorcnt(0);
  }
  __syncthreads();

  const int w    = threadIdx.x >> 5;
  const int lane = threadIdx.x & 31;
  const int col  = lane & 15;
  const int hi   = lane >> 4;

  v8f acc0 = {0.f,0.f,0.f,0.f,0.f,0.f,0.f,0.f};
  v8f acc1 = acc0, acc2 = acc0;

  const bf16_t* aRow  = sA + (size_t)(w * 16 + col) * K2_;
  const bf16_t* bRow0 = sB + (size_t)(0 * 16 + col) * K2_;
  const bf16_t* bRow1 = sB + (size_t)(1 * 16 + col) * K2_ + 16 * K2_ * 0; // group1 region
  bRow1 = sB + (size_t)(16 + col) * K2_;
  const bf16_t* bRow2 = sB + (size_t)(32 + col) * K2_;

  for (int kc = 0; kc < K2_ / 32; ++kc) {
    const int k0 = kc * 32;
    v16bf a  = load_a_frag(aRow, k0, hi);
    v16bf b0 = load_b_frag(bRow0, k0, hi);
    v16bf b1 = load_b_frag(bRow1, k0, hi);
    v16bf b2 = load_b_frag(bRow2, k0, hi);
    acc0 = __builtin_amdgcn_wmma_f32_16x16x32_bf16(false, a, false, b0, (short)0, acc0, false, false);
    acc1 = __builtin_amdgcn_wmma_f32_16x16x32_bf16(false, a, false, b1, (short)0, acc1, false, false);
    acc2 = __builtin_amdgcn_wmma_f32_16x16x32_bf16(false, a, false, b2, (short)0, acc2, false, false);
  }

  const int n  = n0 + col;
  const float bi = b_hid[n];
  const float bf = b_hid[OD_ + n];
  const float bh = b_hid[2 * OD_ + n];
#pragma unroll
  for (int r = 0; r < 8; ++r) {
    size_t m  = (size_t)m0 + w * 16 + r + hi * 8;   // C layout: M = r + 8*(lane>=16)
    float ig = acc0[r] + bi;
    float fg = acc1[r] + bf;
    float hd = acc2[r] + bh;
    fgate[m * OD_ + n] = sigmoidf_(fg);
    igh[m * OD_ + n]   = sigmoidf_(ig) * fmaxf(hd, 0.0f);
  }
}

// ================= K3: recurrence c = f*c + igh along S =====================
__global__ void k_recurrence(const float* __restrict__ fgate,
                             const float* __restrict__ igh,
                             const float* __restrict__ init_cx,
                             float* __restrict__ cell,
                             bf16_t* __restrict__ x2 /* high half becomes cell(bf16) */) {
  int t = blockIdx.x * blockDim.x + threadIdx.x;    // B*H*OD = 16384 lanes
  int b = t / (H_ * OD_);
  int r = t % (H_ * OD_);
  int h = r / OD_, od = r % OD_;
  float c = init_cx[h * OD_ + od];
  for (int s = 0; s < S_; ++s) {
    size_t row = ((size_t)b * S_ + s) * H_ + h;
    size_t idx = row * OD_ + od;
    if (s + 1 < S_) {
      __builtin_prefetch(fgate + idx + (size_t)H_ * OD_, 0, 1);
      __builtin_prefetch(igh   + idx + (size_t)H_ * OD_, 0, 1);
    }
    c = fgate[idx] * c + igh[idx];
    cell[idx] = c;
    x2[row * K2_ + ID_ + od] = (bf16_t)c;           // xo = [hin | cell]
  }
}

// ================= K4: GEMM2 (TDM-staged, bf16 WMMA) + output ===============
// block = 128 threads (4 waves). M tile = 64 rows, N tile = 64 cols (4 x 16).
__global__ void k_gemm2_out(const bf16_t* __restrict__ xo,
                            const bf16_t* __restrict__ wog,
                            const float* __restrict__ b_og,
                            const float* __restrict__ cell,
                            float* __restrict__ out) {
  extern __shared__ char smem_raw[];
  bf16_t* sA = (bf16_t*)smem_raw;                      // 64 x 512 (64 KiB)
  bf16_t* sB = (bf16_t*)(smem_raw + 64 * K2_ * 2);     // 64 x 512 (64 KiB)

  const int m0 = blockIdx.x * 64;
  const int n0 = blockIdx.y * 64;

  if (threadIdx.x < 32) {
    tdm_load_2d(lds_off_(sA), xo + (size_t)m0 * K2_, K2_, 64, K2_, M_, K2_, 1);
    tdm_load_2d(lds_off_(sB), wog + (size_t)n0 * K2_, K2_, 64, K2_, OD_, K2_, 1);
    __builtin_amdgcn_s_wait_tensorcnt(0);
  }
  __syncthreads();

  const int w    = threadIdx.x >> 5;
  const int lane = threadIdx.x & 31;
  const int col  = lane & 15;
  const int hi   = lane >> 4;

  v8f acc[4];
#pragma unroll
  for (int tI = 0; tI < 4; ++tI) acc[tI] = (v8f){0.f,0.f,0.f,0.f,0.f,0.f,0.f,0.f};

  const bf16_t* aRow = sA + (size_t)(w * 16 + col) * K2_;
  for (int kc = 0; kc < K2_ / 32; ++kc) {
    const int k0 = kc * 32;
    v16bf a = load_a_frag(aRow, k0, hi);
#pragma unroll
    for (int tI = 0; tI < 4; ++tI) {
      const bf16_t* bRow = sB + (size_t)(tI * 16 + col) * K2_;
      v16bf b = load_b_frag(bRow, k0, hi);
      acc[tI] = __builtin_amdgcn_wmma_f32_16x16x32_bf16(false, a, false, b, (short)0, acc[tI], false, false);
    }
  }

#pragma unroll
  for (int tI = 0; tI < 4; ++tI) {
    const int n = n0 + tI * 16 + col;
    const float bb = b_og[n];
#pragma unroll
    for (int r = 0; r < 8; ++r) {
      size_t m = (size_t)m0 + w * 16 + r + hi * 8;
      float og = sigmoidf_(acc[tI][r] + bb);
      float cv = cell[m * OD_ + n];
      out[m * OD_ + n] = og * cv;
    }
  }
}

// ================= host launcher =================
extern "C" void kernel_launch(void* const* d_in, const int* in_sizes, int n_in,
                              void* d_out, int out_size, void* d_ws, size_t ws_size,
                              hipStream_t stream) {
  const float* hin    = (const float*)d_in[0];  // (B,S,H,ID)
  const float* Whid   = (const float*)d_in[1];  // (768,512)
  const float* bhid   = (const float*)d_in[2];  // (768,)
  const float* Wog    = (const float*)d_in[3];  // (256,512)
  const float* bog    = (const float*)d_in[4];  // (256,)
  const float* ln_w   = (const float*)d_in[5];  // (H,ID)
  const float* ln_b   = (const float*)d_in[6];  // (H,ID)
  const float* initcx = (const float*)d_in[7];  // (H,OD)
  float* out = (float*)d_out;

  char* w = (char*)d_ws;
  size_t off = 0;
  bf16_t* x2   = (bf16_t*)(w + off); off += (size_t)M_ * K2_ * sizeof(bf16_t); // 128 MiB
  float*  fg   = (float*) (w + off); off += (size_t)M_ * OD_ * sizeof(float);  // 128 MiB
  float*  ig   = (float*) (w + off); off += (size_t)M_ * OD_ * sizeof(float);  // 128 MiB
  float*  cell = (float*) (w + off); off += (size_t)M_ * OD_ * sizeof(float);  // 128 MiB
  bf16_t* whb  = (bf16_t*)(w + off); off += (size_t)N1_ * K2_ * sizeof(bf16_t);
  bf16_t* wob  = (bf16_t*)(w + off); off += (size_t)OD_ * K2_ * sizeof(bf16_t);
  (void)ws_size; (void)out_size; (void)in_sizes; (void)n_in;

  // K0: weight conversion
  {
    int n = N1_ * K2_;
    k_convert_weights<<<(n + 255) / 256, 256, 0, stream>>>(Whid, Wog, whb, wob);
  }
  // K1a: cumsum + pack bf16 x2
  k_cumsum_pack<<<(B_ * H_ * ID_) / 256, 256, 0, stream>>>(hin, x2);
  // K1b: LayerNorm per (b,s)
  k_layernorm<<<B_ * S_, 256, 0, stream>>>(x2, ln_w, ln_b);
  // K2: GEMM1 + gates
  {
    dim3 grid(M_ / 64, OD_ / 16);
    size_t smem = (size_t)64 * K2_ * 2 + (size_t)3 * 16 * K2_ * 2;  // 112 KiB
    k_gemm1_gates<<<grid, 128, smem, stream>>>(x2, whb, bhid, fg, ig);
  }
  // K3: recurrence
  k_recurrence<<<(B_ * H_ * OD_) / 256, 256, 0, stream>>>(fg, ig, initcx, cell, x2);
  // K4: GEMM2 + output
  {
    dim3 grid(M_ / 64, OD_ / 64);
    size_t smem = (size_t)64 * K2_ * 2 * 2;                         // 128 KiB
    k_gemm2_out<<<grid, 128, smem, stream>>>(x2, wob, bog, cell, out);
  }
}
